// FERNet_82480551953143
// MI455X (gfx1250) — matrix operand order repaired and checked by reading, not compiled, not run on hardware
//
#include <hip/hip_runtime.h>
#include <hip/hip_bf16.h>

typedef __attribute__((ext_vector_type(16))) __bf16 bf16x16;
typedef __attribute__((ext_vector_type(8)))  float  f32x8;

// ---------------------------------------------------------------------------
// Core: implicit-GEMM 3x3 SAME conv, bf16 in, f32 accumulate via WMMA.
// X: [N,H+2,W+2,Cin_pad] bf16 with 1-pixel ZERO HALO (Cin_pad % 32 == 0).
// Wf: fragment-packed weights [cb][tap][kb][lane][16] bf16.
// Each wave computes a 2x2 register tile: 2 cout tiles x 2 pixel strips
// (32 couts x 32 pixels). Per K-block: 2 A-loads + 2 B-loads feed 4 WMMAs
// -> 2 global_load_b128 per WMMA, no boundary predication.
// grid(W/128, H, N*Cout_pad/32), block 128 (4 waves, each 32 pixels).
// ---------------------------------------------------------------------------
__global__ __launch_bounds__(128)
void conv3x3_wmma(const __hip_bfloat16* __restrict__ X,
                  const __hip_bfloat16* __restrict__ Wf,
                  const float* __restrict__ bias,
                  __hip_bfloat16* __restrict__ Ybf,
                  float* __restrict__ Yf32,
                  const __hip_bfloat16* __restrict__ R,
                  int H, int Wd, int Cin_pad, int Cout_pad,
                  int relu, int f32out, int haloOut)
{
  const int lane  = threadIdx.x & 31;
  const int wave  = threadIdx.x >> 5;
  const int nPair = Cout_pad >> 5;
  const int pr    = blockIdx.z % nPair;
  const int n     = blockIdx.z / nPair;
  const int h     = blockIdx.y;
  const int w0    = (blockIdx.x * 4 + wave) << 5;   // 32 pixels per wave
  const int nKb   = Cin_pad >> 5;
  const int cb0   = pr * 2;
  const int Hs = H + 2, Ws = Wd + 2;

  const int px   = lane & 15;          // B column = pixel within strip
  const int kh16 = (lane >> 4) << 4;   // K offset: lanes 16-31 hold K=16..31
  const int mOff = (lane >> 4) << 3;   // C/D row offset: lanes 16-31 -> M+8

  f32x8 acc[2][2];
#pragma unroll
  for (int t = 0; t < 2; ++t)
#pragma unroll
    for (int s = 0; s < 2; ++s)
#pragma unroll
      for (int i = 0; i < 8; ++i) acc[t][s][i] = bias[cb0 * 16 + t * 16 + mOff + i];

  const long laneW    = (long)lane * 16;
  const long cbStride = (long)9 * nKb * 512;
  const long stripStr = (long)16 * Cin_pad;         // 16 pixels in elements
  for (int tap = 0; tap < 9; ++tap) {
    const int dy = tap / 3 - 1, dx = tap % 3 - 1;
    // halo layout: logical (hh,ww) lives at (hh+1, ww+1); always in-bounds.
    const __hip_bfloat16* xp =
        X + ((long)(n * Hs + h + 1 + dy) * Ws + (w0 + px + 1 + dx)) * Cin_pad + kh16;
    const __hip_bfloat16* wp = Wf + ((long)(cb0 * 9 + tap) * nKb) * 512 + laneW;
    for (int kb = 0; kb < nKb; ++kb) {
      bf16x16 b0 = *(const bf16x16*)(xp + (long)kb * 32);
      bf16x16 b1 = *(const bf16x16*)(xp + stripStr + (long)kb * 32);
#pragma unroll
      for (int t = 0; t < 2; ++t) {
        bf16x16 a = *(const bf16x16*)(wp + (long)t * cbStride + (long)kb * 512);
        acc[t][0] = __builtin_amdgcn_wmma_f32_16x16x32_bf16(false, a, false, b0,
                                                            (short)0, acc[t][0], false, false);
        acc[t][1] = __builtin_amdgcn_wmma_f32_16x16x32_bf16(false, a, false, b1,
                                                            (short)0, acc[t][1], false, false);
      }
    }
  }

#pragma unroll
  for (int s = 0; s < 2; ++s) {
    const int wpix = w0 + s * 16 + px;
    const long pixBase = haloOut ? ((long)(n * Hs + h + 1) * Ws + (wpix + 1))
                                 : ((long)(n * H + h) * Wd + wpix);
#pragma unroll
    for (int t = 0; t < 2; ++t) {
      const long cOff = pixBase * Cout_pad + cb0 * 16 + t * 16 + mOff;
      if (R) {
        union { uint4 u; __hip_bfloat16 hh[8]; } ru;
        ru.u = *(const uint4*)(R + cOff);
#pragma unroll
        for (int i = 0; i < 8; ++i) acc[t][s][i] += __bfloat162float(ru.hh[i]);
      }
      if (relu) {
#pragma unroll
        for (int i = 0; i < 8; ++i) acc[t][s][i] = fmaxf(acc[t][s][i], 0.0f);
      }
      if (f32out) {
        *(f32x8*)(Yf32 + cOff) = acc[t][s];
      } else {
        union { uint4 u; __hip_bfloat162 h2[4]; } pk;
#pragma unroll
        for (int i = 0; i < 4; ++i)
          pk.h2[i] = __float22bfloat162_rn(make_float2(acc[t][s][2 * i], acc[t][s][2 * i + 1]));
        *(uint4*)(Ybf + cOff) = pk.u;
      }
    }
  }
}

// ---------------------------------------------------------------------------
// Weight-norm scale: scale[co] = g[co] / sqrt(sum v[co,:]^2). One block/cout.
// ---------------------------------------------------------------------------
__global__ void scale_kernel(const float* __restrict__ v, const float* __restrict__ g,
                             float* __restrict__ scale, int per, int Cout)
{
  __shared__ float red[64];
  int co = blockIdx.x;
  if (co >= Cout) return;
  float s = 0.f;
  for (int i = threadIdx.x; i < per; i += 64) {
    float t = v[(long)co * per + i];
    s += t * t;
  }
  red[threadIdx.x] = s;
  __syncthreads();
  for (int st = 32; st > 0; st >>= 1) {
    if ((int)threadIdx.x < st) red[threadIdx.x] += red[threadIdx.x + st];
    __syncthreads();
  }
  if (threadIdx.x == 0) scale[co] = g[co] / sqrtf(red[0]);
}

// ---------------------------------------------------------------------------
// Pack OIHW f32 weights into WMMA A-fragment order, applying optional scale.
// Layout: [cb][tap][kb][lane][e] ; K = 16*(lane/16) + e (contiguous, same order as the B loads).
// ---------------------------------------------------------------------------
__global__ void pack_w_kernel(const float* __restrict__ v, const float* __restrict__ scale,
                              __hip_bfloat16* __restrict__ Wf,
                              int Cin, int Cout, int Cin_pad, int Cout_pad, long total)
{
  const int nKb = Cin_pad >> 5;
  for (long idx = (long)blockIdx.x * blockDim.x + threadIdx.x; idx < total;
       idx += (long)gridDim.x * blockDim.x) {
    int e    = (int)(idx & 15);
    int lane = (int)((idx >> 4) & 31);
    long rest = idx >> 9;
    int kb  = (int)(rest % nKb); rest /= nKb;
    int tap = (int)(rest % 9);
    int cb  = (int)(rest / 9);
    int m = lane & 15;
    int K = ((lane >> 4) << 4) + e;
    int cout = cb * 16 + m;
    int ci   = kb * 32 + K;
    float val = 0.f;
    if (cout < Cout && ci < Cin) {
      float s = scale ? scale[cout] : 1.f;
      val = v[(((long)cout * Cin + ci) * 3 + tap / 3) * 3 + tap % 3] * s;
    }
    Wf[idx] = __float2bfloat16(val);
  }
}

__global__ void pack_bias_kernel(const float* __restrict__ b, float* __restrict__ bo,
                                 int Cout, int Cout_pad)
{
  int i = threadIdx.x;
  if (i < Cout_pad) bo[i] = (i < Cout) ? b[i] : 0.f;
}

// ---------------------------------------------------------------------------
// Zero the 1-pixel halo ring of a [N,H+2,W+2,Cpad] bf16 buffer.
// ---------------------------------------------------------------------------
__global__ void zero_halo_kernel(__hip_bfloat16* __restrict__ buf,
                                 int H, int Wd, int Cpad, long total)
{
  const int Hs = H + 2, Ws = Wd + 2;
  const int ring = 2 * Ws + 2 * H;
  for (long idx = (long)blockIdx.x * blockDim.x + threadIdx.x; idx < total;
       idx += (long)gridDim.x * blockDim.x) {
    int c = (int)(idx % Cpad); long p = idx / Cpad;
    int k = (int)(p % ring);
    int n = (int)(p / ring);
    int row, col;
    if (k < Ws)            { row = 0;      col = k; }
    else if (k < 2 * Ws)   { row = Hs - 1; col = k - Ws; }
    else { int k2 = k - 2 * Ws; row = 1 + (k2 >> 1); col = (k2 & 1) ? (Ws - 1) : 0; }
    buf[((long)(n * Hs + row) * Ws + col) * Cpad + c] = __float2bfloat16(0.f);
  }
}

// ---------------------------------------------------------------------------
// Tiny scalar KPN (one thread): coord -> (3,6,3,3) kernel + bias(3, padded 32)
// ---------------------------------------------------------------------------
__global__ void kpn_kernel(const float* __restrict__ pw, const float* __restrict__ pb,
                           const float* __restrict__ kw, const float* __restrict__ kb,
                           const float* __restrict__ bw, const float* __restrict__ bb,
                           float* __restrict__ outW, float* __restrict__ outB,
                           int cx, int cy)
{
  if (threadIdx.x != 0 || blockIdx.x != 0) return;
  float fx = (float)(cx - 4), fy = (float)(cy - 4);
  float h[9], t[9];
  for (int i = 0; i < 9; ++i) h[i] = pw[i * 2] * fx + pw[i * 2 + 1] * fy + pb[i];
  for (int L = 0; L < 18; ++L) {
    if (L) for (int i = 0; i < 9; ++i) h[i] = fmaxf(h[i], 0.f);
    for (int j = 0; j < 9; ++j) {
      float a = kb[L * 9 + j];
      for (int i = 0; i < 9; ++i) a += h[i] * kw[((long)L * 9 + j) * 9 + i];
      t[j] = a;
    }
    for (int j = 0; j < 9; ++j) { h[j] = t[j]; outW[L * 9 + j] = t[j]; }
  }
  for (int k = 0; k < 32; ++k) {
    float a = 0.f;
    if (k < 3) {
      a = bb[k];
      for (int j = 0; j < 9; ++j) a += fmaxf(h[j], 0.f) * bw[k * 9 + j];
    }
    outB[k] = a;
  }
}

// ---------------------------------------------------------------------------
// Elementwise gather / pack kernels. Halo-padded targets write zeros on halo.
// ---------------------------------------------------------------------------
__global__ void pack_inp_kernel(const float* __restrict__ x, __hip_bfloat16* __restrict__ out,
                                long total)  // [N,258,258,96], ch<81 = x[n,r1,r2,1,h,w]
{
  for (long idx = (long)blockIdx.x * blockDim.x + threadIdx.x; idx < total;
       idx += (long)gridDim.x * blockDim.x) {
    int c = (int)(idx % 96); long p = idx / 96;
    int ws = (int)(p % 258); p /= 258;
    int hs = (int)(p % 258);
    int n  = (int)(p / 258);
    int h = hs - 1, w = ws - 1;
    float v = 0.f;
    if (c < 81 && h >= 0 && h < 256 && w >= 0 && w < 256) {
      int r1 = c / 9, r2 = c % 9;
      v = x[(((((long)n * 9 + r1) * 9 + r2) * 3 + 1) * 256 + h) * 256 + w];
    }
    out[idx] = __float2bfloat16(v);
  }
}

__global__ void pack_comb_kernel(const __hip_bfloat16* __restrict__ GR,  // flat [N,256,256,32]
                                 const float* __restrict__ x,
                                 __hip_bfloat16* __restrict__ out,       // halo [N,258,258,32]
                                 int cx, int cy, long total)
{
  for (long idx = (long)blockIdx.x * blockDim.x + threadIdx.x; idx < total;
       idx += (long)gridDim.x * blockDim.x) {
    int c = (int)(idx % 32); long p = idx / 32;
    int ws = (int)(p % 258); p /= 258;
    int hs = (int)(p % 258);
    int n  = (int)(p / 258);
    int h = hs - 1, w = ws - 1;
    __hip_bfloat16 v = __float2bfloat16(0.f);
    if (h >= 0 && h < 256 && w >= 0 && w < 256) {
      long q = (long)(n * 256 + h) * 256 + w;
      if (c < 3) v = GR[q * 32 + c];
      else if (c < 6)
        v = __float2bfloat16(x[(((((long)n * 9 + cx) * 9 + cy) * 3 + (c - 3)) * 256 + h) * 256 + w]);
    }
    out[idx] = v;
  }
}

__global__ void pack_ffn_kernel(const float* __restrict__ x,
                                const __hip_bfloat16* __restrict__ GR,   // flat *32
                                const __hip_bfloat16* __restrict__ CR,   // flat *32
                                __hip_bfloat16* __restrict__ out,        // halo [N,258,258,64]
                                int cx, int cy, long total)
{
  for (long idx = (long)blockIdx.x * blockDim.x + threadIdx.x; idx < total;
       idx += (long)gridDim.x * blockDim.x) {
    int c = (int)(idx % 64); long p = idx / 64;
    int ws = (int)(p % 258); p /= 258;
    int hs = (int)(p % 258);
    int n  = (int)(p / 258);
    int h = hs - 1, w = ws - 1;
    __hip_bfloat16 v = __float2bfloat16(0.f);
    if (h >= 0 && h < 256 && w >= 0 && w < 256) {
      long q = (long)(n * 256 + h) * 256 + w;
      if (c < 27) {
        int i = c / 9, j = (c % 9) / 3, cc = c % 3;
        v = __float2bfloat16(
            x[(((((long)n * 9 + (cx - 1 + i)) * 9 + (cy - 1 + j)) * 3 + cc) * 256 + h) * 256 + w]);
      } else if (c < 30) v = CR[q * 32 + (c - 27)];
      else if (c < 33)   v = GR[q * 32 + (c - 30)];
    }
    out[idx] = v;
  }
}

__global__ void make_hs_kernel(const __hip_bfloat16* __restrict__ CS,   // halo [N,258,258,64]
                               const __hip_bfloat16* __restrict__ F0,   // halo [N,258,258,64]
                               __hip_bfloat16* __restrict__ out,        // halo [N,130,130,288]
                               long total)
{  // concat(unshuffle(cs)(132), unshuffle(x)(132)), pad 0
  for (long idx = (long)blockIdx.x * blockDim.x + threadIdx.x; idx < total;
       idx += (long)gridDim.x * blockDim.x) {
    int c = (int)(idx % 288); long p = idx / 288;
    int ws = (int)(p % 130); p /= 130;
    int hs = (int)(p % 130);
    int n  = (int)(p / 130);
    int ho = hs - 1, wo = ws - 1;
    __hip_bfloat16 v = __float2bfloat16(0.f);
    if (c < 264 && ho >= 0 && ho < 128 && wo >= 0 && wo < 128) {
      const __hip_bfloat16* src = (c < 132) ? CS : F0;
      int cl = (c < 132) ? c : c - 132;
      int ch = cl >> 2, dy = (cl >> 1) & 1, dx = cl & 1;
      v = src[((long)(n * 258 + (ho * 2 + dy) + 1) * 258 + (wo * 2 + dx) + 1) * 64 + ch];
    }
    out[idx] = v;
  }
}

__global__ void make_y_kernel(const __hip_bfloat16* __restrict__ HS,   // halo [N,130,130,288]
                              const __hip_bfloat16* __restrict__ CS,   // halo [N,258,258,64]
                              const __hip_bfloat16* __restrict__ F0,   // halo [N,258,258,64]
                              __hip_bfloat16* __restrict__ out,        // halo [N,258,258,160]
                              long total)
{  // concat(shuffle(hs)(66), cs(33), x(33)), pad 0
  for (long idx = (long)blockIdx.x * blockDim.x + threadIdx.x; idx < total;
       idx += (long)gridDim.x * blockDim.x) {
    int c = (int)(idx % 160); long p = idx / 160;
    int ws = (int)(p % 258); p /= 258;
    int hs = (int)(p % 258);
    int n  = (int)(p / 258);
    int h = hs - 1, w = ws - 1;
    __hip_bfloat16 v = __float2bfloat16(0.f);
    if (h >= 0 && h < 256 && w >= 0 && w < 256) {
      if (c < 66) {
        int ic = c * 4 + (h & 1) * 2 + (w & 1);
        v = HS[((long)(n * 130 + (h >> 1) + 1) * 130 + (w >> 1) + 1) * 288 + ic];
      } else if (c < 99) {
        v = CS[((long)(n * 258 + h + 1) * 258 + w + 1) * 64 + (c - 66)];
      } else if (c < 132) {
        v = F0[((long)(n * 258 + h + 1) * 258 + w + 1) * 64 + (c - 99)];
      }
    }
    out[idx] = v;
  }
}

__global__ void scatter_kernel(const float* __restrict__ O32, float* __restrict__ out,
                               int cx, int cy, long total)
{  // O32: flat [N,256,256,32] f32, ch = i*9+j*3+cc -> out[n,cx-1+i,cy-1+j,cc,h,w], clipped
  for (long idx = (long)blockIdx.x * blockDim.x + threadIdx.x; idx < total;
       idx += (long)gridDim.x * blockDim.x) {
    int w = (int)(idx % 256); long p = idx / 256;
    int h = (int)(p % 256); p /= 256;
    int c = (int)(p % 27);
    int n = (int)(p / 27);
    float v = O32[((long)(n * 256 + h) * 256 + w) * 32 + c];
    v = fminf(fmaxf(v, 0.f), 1.f);
    int i = c / 9, j = (c % 9) / 3, cc = c % 3;
    out[(((((long)n * 9 + (cx - 1 + i)) * 9 + (cy - 1 + j)) * 3 + cc) * 256 + h) * 256 + w] = v;
  }
}

// ---------------------------------------------------------------------------
// Host orchestration
// ---------------------------------------------------------------------------
struct Layer {
  const __hip_bfloat16* frag;
  const float* bias;
  int Cin_pad, Cout_pad;
};

static inline unsigned gblocks(long n) {
  long b = (n + 255) / 256;
  if (b > 1048576) b = 1048576;
  return (unsigned)b;
}

extern "C" void kernel_launch(void* const* d_in, const int* in_sizes, int n_in,
                              void* d_out, int out_size, void* d_ws, size_t ws_size,
                              hipStream_t stream)
{
  (void)in_sizes; (void)n_in; (void)out_size; (void)ws_size;

  const float* x   = (const float*)d_in[0];
  const float* e1v = (const float*)d_in[1];
  const float* e1g = (const float*)d_in[2];
  const float* e1b = (const float*)d_in[3];
  const float* e2v = (const float*)d_in[4];
  const float* e2g = (const float*)d_in[5];
  const float* e2b = (const float*)d_in[6];
  const float* pw  = (const float*)d_in[7];
  const float* pb  = (const float*)d_in[8];
  const float* kwv = (const float*)d_in[9];
  const float* kbv = (const float*)d_in[10];
  const float* bwv = (const float*)d_in[11];
  const float* bbv = (const float*)d_in[12];

  const int N = 2;
  const long HW = 65536;          // 256*256 (flat buffers)
  const long P  = 66564;          // 258*258 (halo buffers)
  const long P2 = 16900;          // 130*130 (halo buffers @128)

  char* ws = (char*)d_ws;
  size_t cur = 0;
  auto alloc = [&](size_t bytes) -> void* {
    cur = (cur + 255) & ~(size_t)255;
    void* p = ws + cur;
    cur += bytes;
    return p;
  };

  // --- activation buffers (layout order matters for aliasing; all sizes are
  //     multiples of 256B so consecutive allocs are contiguous) ---
  __hip_bfloat16* INP = (__hip_bfloat16*)alloc((size_t)N * P * 96 * 2);    // 25.56 MB (halo)
  __hip_bfloat16* G64 = (__hip_bfloat16*)alloc((size_t)N * P * 64 * 2);    // 17.04 MB (halo)
  __hip_bfloat16* GR  = (__hip_bfloat16*)alloc((size_t)N * HW * 32 * 2);   // flat
  __hip_bfloat16* CMB = (__hip_bfloat16*)alloc((size_t)N * P * 32 * 2);    // halo
  __hip_bfloat16* CR  = (__hip_bfloat16*)alloc((size_t)N * HW * 32 * 2);   // flat
  __hip_bfloat16* F0  = (__hip_bfloat16*)alloc((size_t)N * P * 64 * 2);    // halo
  __hip_bfloat16* T1  = (__hip_bfloat16*)alloc((size_t)N * P * 64 * 2);    // halo
  __hip_bfloat16* CS  = (__hip_bfloat16*)alloc((size_t)N * P * 64 * 2);    // halo
  __hip_bfloat16* Hb0 = (__hip_bfloat16*)alloc((size_t)N * P2 * 288 * 2);  // halo @128
  __hip_bfloat16* Hb1 = (__hip_bfloat16*)alloc((size_t)N * P2 * 288 * 2);
  __hip_bfloat16* Hb2 = (__hip_bfloat16*)alloc((size_t)N * P2 * 288 * 2);
  float*          O32 = (float*)alloc((size_t)N * HW * 32 * 4);            // flat f32
  // aliases over dead regions:
  __hip_bfloat16* Y0 = INP;  // needs N*P*160*2 == sizeof(INP)+sizeof(G64) exactly
  __hip_bfloat16* Y1 = Hb0;  // Hb region (3x19.47MB) dead once Y phase begins
  __hip_bfloat16* Y2 = F0;   // F0+T1+CS (51.1MB) dead once Y2 written

  float* scaleBuf = (float*)alloc(512 * 4);
  float* kpnW     = (float*)alloc(256 * 4);
  float* kpnB     = (float*)alloc(32 * 4);
  __hip_bfloat16* kpnFrag = (__hip_bfloat16*)alloc((size_t)2 * 9 * 512 * 2);

  auto mk_layer = [&](const float* v, const float* g, const float* b,
                      int Cin, int Cout, int Cin_pad, int Cout_pad) -> Layer {
    size_t fragElems = (size_t)(Cout_pad / 16) * 9 * (Cin_pad / 32) * 512;
    __hip_bfloat16* frag = (__hip_bfloat16*)alloc(fragElems * 2);
    float* bias = (float*)alloc((size_t)Cout_pad * 4);
    scale_kernel<<<Cout, 64, 0, stream>>>(v, g, scaleBuf, Cin * 9, Cout);
    pack_w_kernel<<<gblocks((long)fragElems), 256, 0, stream>>>(
        v, scaleBuf, frag, Cin, Cout, Cin_pad, Cout_pad, (long)fragElems);
    pack_bias_kernel<<<1, Cout_pad, 0, stream>>>(b, bias, Cout, Cout_pad);
    return Layer{frag, bias, Cin_pad, Cout_pad};
  };

  auto conv = [&](const __hip_bfloat16* X, const Layer& L, __hip_bfloat16* Ybf,
                  float* Yf32, const __hip_bfloat16* R, int Hh, int Ww,
                  int relu, int f32out, int haloOut) {
    dim3 grid((unsigned)(Ww / 128), (unsigned)Hh, (unsigned)(N * (L.Cout_pad / 32)));
    conv3x3_wmma<<<grid, 128, 0, stream>>>(X, L.frag, L.bias, Ybf, Yf32, R,
                                           Hh, Ww, L.Cin_pad, L.Cout_pad, relu, f32out, haloOut);
  };

  auto zhalo = [&](__hip_bfloat16* buf, int Hh, int Ww, int Cpad) {
    long t = (long)N * (2 * (Ww + 2) + 2 * Hh) * Cpad;
    zero_halo_kernel<<<gblocks(t), 256, 0, stream>>>(buf, Hh, Ww, Cpad, t);
  };

  // --- pack all static conv weights (weight-norm + fragment order) ---
  Layer Le1  = mk_layer(e1v, e1g, e1b, 81, 64, 96, 64);
  Layer Le2  = mk_layer(e2v, e2g, e2b, 64, 3, 64, 32);
  Layer Lr1a = mk_layer((const float*)d_in[13], (const float*)d_in[14], (const float*)d_in[15], 33, 33, 64, 64);
  Layer Lr1b = mk_layer((const float*)d_in[16], (const float*)d_in[17], (const float*)d_in[18], 33, 33, 64, 64);
  Layer Lr2a = mk_layer((const float*)d_in[19], (const float*)d_in[20], (const float*)d_in[21], 264, 264, 288, 288);
  Layer Lr2b = mk_layer((const float*)d_in[22], (const float*)d_in[23], (const float*)d_in[24], 264, 264, 288, 288);
  Layer Lr2c = mk_layer((const float*)d_in[25], (const float*)d_in[26], (const float*)d_in[27], 264, 264, 288, 288);
  Layer Lr2d = mk_layer((const float*)d_in[28], (const float*)d_in[29], (const float*)d_in[30], 264, 264, 288, 288);
  Layer Lora = mk_layer((const float*)d_in[31], (const float*)d_in[32], (const float*)d_in[33], 132, 132, 160, 160);
  Layer Lorb = mk_layer((const float*)d_in[34], (const float*)d_in[35], (const float*)d_in[36], 132, 132, 160, 160);
  Layer Loc  = mk_layer((const float*)d_in[37], (const float*)d_in[38], (const float*)d_in[39], 132, 27, 160, 32);

  // --- embedding: inp -> g -> grepr (computed once, persists across centers) ---
  {
    long t = (long)N * P * 96;
    pack_inp_kernel<<<gblocks(t), 256, 0, stream>>>(x, INP, t);
  }
  zhalo(G64, 256, 256, 64);
  conv(INP, Le1, G64, nullptr, nullptr, 256, 256, 1, 0, 1);
  conv(G64, Le2, GR, nullptr, nullptr, 256, 256, 1, 0, 0);

  // --- 9 centers (disjoint output patches; cnt == 1 everywhere) ---
  const int centers[3] = {1, 4, 7};
  for (int a = 0; a < 3; ++a) {
    for (int b2 = 0; b2 < 3; ++b2) {
      const int cx = centers[a], cy = centers[b2];

      // dynamic KPN kernel + pack (Cout_pad 32)
      kpn_kernel<<<1, 32, 0, stream>>>(pw, pb, kwv, kbv, bwv, bbv, kpnW, kpnB, cx, cy);
      pack_w_kernel<<<gblocks(2 * 9 * 512), 256, 0, stream>>>(
          kpnW, nullptr, kpnFrag, 6, 3, 32, 32, (long)(2 * 9 * 512));
      Layer Lk{kpnFrag, kpnB, 32, 32};

      // crepr = conv(concat(grepr, x[cx,cy]), kpn)
      {
        long t = (long)N * P * 32;
        pack_comb_kernel<<<gblocks(t), 256, 0, stream>>>(GR, x, CMB, cx, cy, t);
      }
      conv(CMB, Lk, CR, nullptr, nullptr, 256, 256, 0, 0, 0);

      // ffn input: neigh(27) + crepr(3) + grepr(3)
      {
        long t = (long)N * P * 64;
        pack_ffn_kernel<<<gblocks(t), 256, 0, stream>>>(x, GR, CR, F0, cx, cy, t);
      }

      // res1
      zhalo(T1, 256, 256, 64);
      conv(F0, Lr1a, T1, nullptr, nullptr, 256, 256, 1, 0, 1);
      conv(T1, Lr1b, CS, nullptr, F0, 256, 256, 0, 0, 1);

      // res2 @128x128 on concat(unshuffle(cs), unshuffle(x))
      {
        long t = (long)N * P2 * 288;
        make_hs_kernel<<<gblocks(t), 256, 0, stream>>>(CS, F0, Hb0, t);
      }
      zhalo(Hb1, 128, 128, 288);
      zhalo(Hb2, 128, 128, 288);
      conv(Hb0, Lr2a, Hb1, nullptr, nullptr, 128, 128, 1, 0, 1);
      conv(Hb1, Lr2b, Hb2, nullptr, nullptr, 128, 128, 1, 0, 1);
      conv(Hb2, Lr2c, Hb1, nullptr, nullptr, 128, 128, 1, 0, 1);
      conv(Hb1, Lr2d, Hb2, nullptr, Hb0, 128, 128, 0, 0, 1);

      // out_res on concat(shuffle(hs), cs, x)
      {
        long t = (long)N * P * 160;
        make_y_kernel<<<gblocks(t), 256, 0, stream>>>(Hb2, CS, F0, Y0, t);
      }
      zhalo(Y1, 256, 256, 160);
      conv(Y0, Lora, Y1, nullptr, nullptr, 256, 256, 1, 0, 1);
      zhalo(Y2, 256, 256, 160);
      conv(Y1, Lorb, Y2, nullptr, Y0, 256, 256, 0, 0, 1);

      // out_conv -> f32 flat, scatter with clip into d_out
      conv(Y2, Loc, nullptr, O32, nullptr, 256, 256, 0, 1, 0);
      {
        long t = (long)N * 27 * HW;
        scatter_kernel<<<gblocks(t), 256, 0, stream>>>(O32, (float*)d_out, cx, cy, t);
      }
    }
  }
}
